// SwinTransformerBlock_34479997453133
// MI455X (gfx1250) — matrix operand order repaired and checked by reading, not compiled, not run on hardware
//
#include <hip/hip_runtime.h>

// ---------------------------------------------------------------------------
// Problem constants (from reference)
// ---------------------------------------------------------------------------
constexpr int B_   = 32;
constexpr int H_   = 56;
constexpr int W_   = 56;
constexpr int L_   = H_ * W_;        // 3136
constexpr int C_   = 256;            // DIM
constexpr int NH_  = 8;
constexpr int WS_  = 7;
constexpr int NTOK_ = 49;
constexpr int P_   = 5;
constexpr int N_   = 54;             // NTOK + P
constexpr int NWIN_ = 64;
constexpr int CE_  = 256;
constexpr int CH_  = 4;
constexpr int CHD_ = 64;
constexpr float SCALE_  = 0.17677669529663687f;   // 32^-0.5
constexpr float CSCALE_ = 0.0625f;                // 256^-0.5

// ---------------------------------------------------------------------------
// WMMA types / helpers (CDNA5 gfx1250, wave32)
// ---------------------------------------------------------------------------
typedef __attribute__((ext_vector_type(16))) __bf16        v16bf;
typedef __attribute__((ext_vector_type(8)))  float         v8f;
typedef __attribute__((ext_vector_type(8)))  unsigned int  u32x8;

__device__ inline unsigned short f2bfbits(float f) {
  return __builtin_bit_cast(unsigned short, (__bf16)f);   // hw f32->bf16 (RNE)
}
__device__ inline unsigned int packbf2(float a, float b) {
  return (unsigned int)f2bfbits(a) | ((unsigned int)f2bfbits(b) << 16);
}
// Build a 16-element bf16 fragment from two 16B LDS chunks (two ds_load_b128)
__device__ inline v16bf frag_ld(const unsigned int* p, int o0, int o1) {
  uint4 x = *(const uint4*)(p + o0);
  uint4 y = *(const uint4*)(p + o1);
  u32x8 t = { x.x, x.y, x.z, x.w, y.x, y.y, y.z, y.w };
  return __builtin_bit_cast(v16bf, t);
}
__device__ inline float gelu_exact(float v) {
  return 0.5f * v * (1.0f + erff(v * 0.70710678118654752f));
}

// ---------------------------------------------------------------------------
// Generic GEMM: out[M,Nout] = act(A[M,K] @ W[Nout,K]^T + bias) (+ residual)
// bf16 WMMA 16x16x32, f32 accumulate. Compile-time ACT / RES -> branch-free
// epilogue. Block = 256 thr (8 waves); macro tile 64(M) x 128(N); each wave
// owns a 32x32 sub-tile -> 4 WMMAs per K-chunk from 4 fragment builds.
// Requirements: K % 32 == 0, Nout % 128 == 0; M ragged (clamped / guarded).
// ---------------------------------------------------------------------------
template <int ACT, bool RES>
__global__ __launch_bounds__(256)
void gemm_wmma_kernel(const float* __restrict__ A, const float* __restrict__ W,
                      const float* __restrict__ bias, const float* __restrict__ residual,
                      float* __restrict__ out, int M, int Nout, int K)
{
  __shared__ unsigned int sA[64][20];    // 64 rows x 32 bf16 (16 dwords) + pad
  __shared__ unsigned int sB[128][20];   // 128 rows x 32 bf16 + pad

  const int tid  = threadIdx.x;
  const int wave = tid >> 5;
  const int lane = tid & 31;
  const int lr   = lane & 15;
  const int hi   = lane >> 4;
  const int wr   = wave >> 2;            // 0..1 : 32-row slab
  const int wc   = wave & 3;             // 0..3 : 32-col slab
  const int rowBase = blockIdx.x * 64;
  const int colBase = blockIdx.y * 128;

  // staging indices: A: 4 thr/row * 8 floats, B: 2 thr/row * 16 floats
  const int ar = tid >> 2, aq = tid & 3;
  const int br = tid >> 1, bh = tid & 1;
  const int arow = min(rowBase + ar, M - 1);           // clamp: branch-free
  const long aoff = (long)arow * K + aq * 8;
  const long boff = (long)(colBase + br) * K + bh * 16;

  v8f acc00 = {}, acc01 = {}, acc10 = {}, acc11 = {};

  for (int kc = 0; kc < K; kc += 32) {
    {
      const float4* src = (const float4*)(A + aoff + kc);
      float4 f0 = src[0], f1 = src[1];
      uint4 u = { packbf2(f0.x, f0.y), packbf2(f0.z, f0.w),
                  packbf2(f1.x, f1.y), packbf2(f1.z, f1.w) };
      *(uint4*)&sA[ar][aq * 4] = u;
    }
    {
      const float4* src = (const float4*)(W + boff + kc);
      float4 f0 = src[0], f1 = src[1], f2 = src[2], f3 = src[3];
      uint4 u0 = { packbf2(f0.x, f0.y), packbf2(f0.z, f0.w),
                   packbf2(f1.x, f1.y), packbf2(f1.z, f1.w) };
      uint4 u1 = { packbf2(f2.x, f2.y), packbf2(f2.z, f2.w),
                   packbf2(f3.x, f3.y), packbf2(f3.z, f3.w) };
      *(uint4*)&sB[br][bh * 8]     = u0;
      *(uint4*)&sB[br][bh * 8 + 4] = u1;
    }
    __syncthreads();
    v16bf a0 = frag_ld(&sA[wr * 32 + lr][0],      hi * 4, 8 + hi * 4);
    v16bf a1 = frag_ld(&sA[wr * 32 + 16 + lr][0], hi * 4, 8 + hi * 4);
    v16bf b0 = frag_ld(&sB[wc * 32 + lr][0],      hi * 4, 8 + hi * 4);
    v16bf b1 = frag_ld(&sB[wc * 32 + 16 + lr][0], hi * 4, 8 + hi * 4);
    acc00 = __builtin_amdgcn_wmma_f32_16x16x32_bf16(false, a0, false, b0, (short)0, acc00, false, false);
    acc01 = __builtin_amdgcn_wmma_f32_16x16x32_bf16(false, a0, false, b1, (short)0, acc01, false, false);
    acc10 = __builtin_amdgcn_wmma_f32_16x16x32_bf16(false, a1, false, b0, (short)0, acc10, false, false);
    acc11 = __builtin_amdgcn_wmma_f32_16x16x32_bf16(false, a1, false, b1, (short)0, acc11, false, false);
    __syncthreads();
  }

  const int m0 = rowBase + wr * 32;
  const int n0 = colBase + wc * 32;
  const bool fullTile = (rowBase + 64 <= M);   // uniform across block
  #pragma unroll
  for (int ti = 0; ti < 2; ++ti) {
    #pragma unroll
    for (int tj = 0; tj < 2; ++tj) {
      v8f acc = (ti == 0) ? (tj == 0 ? acc00 : acc01) : (tj == 0 ? acc10 : acc11);
      const int n = n0 + tj * 16 + lr;
      const float bn = bias[n];
      const int mstart = m0 + ti * 16 + 8 * hi;
      float* op = out + (size_t)mstart * Nout + n;
      const float* rp = RES ? (residual + (size_t)mstart * Nout + n) : nullptr;
      if (fullTile) {
        #pragma unroll
        for (int r = 0; r < 8; ++r) {
          float v = acc[r] + bn;
          if (ACT == 1) v = gelu_exact(v);
          if (RES) v += rp[(size_t)r * Nout];
          op[(size_t)r * Nout] = v;
        }
      } else {
        #pragma unroll
        for (int r = 0; r < 8; ++r) {
          if (mstart + r < M) {
            float v = acc[r] + bn;
            if (ACT == 1) v = gelu_exact(v);
            if (RES) v += rp[(size_t)r * Nout];
            op[(size_t)r * Nout] = v;
          }
        }
      }
    }
  }
}

// ---------------------------------------------------------------------------
// LayerNorm over last dim (=256). One block per row.
// ---------------------------------------------------------------------------
__global__ __launch_bounds__(256)
void ln_rows_kernel(const float* __restrict__ in, float* __restrict__ out,
                    const float* __restrict__ g, const float* __restrict__ b,
                    int rows)
{
  __shared__ float red[256];
  int row = blockIdx.x;
  int c = threadIdx.x;
  float x = in[(size_t)row * 256 + c];
  red[c] = x; __syncthreads();
  for (int s = 128; s > 0; s >>= 1) { if (c < s) red[c] += red[c + s]; __syncthreads(); }
  float mean = red[0] * (1.0f / 256.0f);
  __syncthreads();
  float d = x - mean;
  red[c] = d * d; __syncthreads();
  for (int s = 128; s > 0; s >>= 1) { if (c < s) red[c] += red[c + s]; __syncthreads(); }
  float var = red[0] * (1.0f / 256.0f);
  out[(size_t)row * 256 + c] = d * rsqrtf(var + 1e-5f) * g[c] + b[c];
}

// ---------------------------------------------------------------------------
// Window gather: xc window rows [P..N) = shifted+partitioned LN(x).
// ---------------------------------------------------------------------------
__global__ void window_gather_kernel(const float* __restrict__ xn, float* __restrict__ xc)
{
  int bid = blockIdx.x;
  int t   = bid % NTOK_; int rem = bid / NTOK_;
  int win = rem % NWIN_; int bb  = rem / NWIN_;
  int wh = win >> 3, ww = win & 7;
  int th = t / WS_, tw = t % WS_;
  int hh = (wh * WS_ + th + 3) % H_;       // cyclic shift by -3 => source +3
  int wv = (ww * WS_ + tw + 3) % W_;
  int l  = hh * W_ + wv;
  int c = threadIdx.x;
  xc[((size_t)(bb * NWIN_ + win) * N_ + P_ + t) * C_ + c] =
      xn[((size_t)bb * L_ + l) * C_ + c];
}

__global__ void fill_spa_kernel(const float* __restrict__ spa, float* __restrict__ xc)
{
  int widx = blockIdx.x;           // B*NWIN
  int bb = widx >> 6;
  int c = threadIdx.x;
  for (int p = 0; p < P_; ++p)
    xc[((size_t)widx * N_ + p) * C_ + c] = spa[(size_t)(bb * P_ + p) * C_ + c];
}

// ---------------------------------------------------------------------------
// Precomputed attention bias table: bm[win][h][m][n] (64x8x64x64 = 8 MB,
// batch-invariant -> stays resident in MI455X's 192 MB L2).
//   n >= 54               : -1e30   (padded keys; forces softmax weight 0)
//   m,n in [5,54)         : relpos bias + shift mask
//   else                  : 0
// ---------------------------------------------------------------------------
__device__ inline int region_val(int r, int c) {
  int rh = (r < 49) ? 0 : ((r < 53) ? 1 : 2);
  int rw = (c < 49) ? 0 : ((c < 53) ? 1 : 2);
  return rh * 3 + rw;
}

__global__ __launch_bounds__(256)
void bias_mask_kernel(const float* __restrict__ rpb, float* __restrict__ bm)
{
  int blk = blockIdx.x;            // win*NH + h
  int h   = blk & 7;
  int win = blk >> 3;
  int wh = win >> 3, ww = win & 7;
  int tid = threadIdx.x;
  #pragma unroll
  for (int i = 0; i < 16; ++i) {
    int idx = i * 256 + tid;       // 0..4095, coalesced
    int m = idx >> 6, n = idx & 63;
    float v = 0.0f;
    if (n >= N_) {
      v = -1e30f;
    } else if (m >= P_ && m < N_ && n >= P_) {
      int ti = m - P_, tj = n - P_;
      int ri = ti / WS_, ci = ti % WS_;
      int rj = tj / WS_, cj = tj % WS_;
      int ridx = (ri - rj + WS_ - 1) * (2 * WS_ - 1) + (ci - cj + WS_ - 1);
      v = rpb[ridx * NH_ + h];
      int vi = region_val(wh * WS_ + ri, ww * WS_ + ci);
      int vj = region_val(wh * WS_ + rj, ww * WS_ + cj);
      if (vi != vj) v -= 100.0f;
    }
    bm[(size_t)blk * 4096 + idx] = v;
  }
}

// ---------------------------------------------------------------------------
// Windowed attention: one block per (window, head). N padded 54->64.
// ---------------------------------------------------------------------------
__global__ __launch_bounds__(256)
void attn_kernel(const float* __restrict__ qkv, const float* __restrict__ bm,
                 float* __restrict__ outbuf)
{
  __shared__ unsigned int sq[64][20];    // Q: 64 tok x 32 d (bf16 packed)
  __shared__ unsigned int sk[64][20];    // K: 64 tok x 32 d
  __shared__ unsigned int svT[32][36];   // V^T: 32 d x 64 tok (bf16 packed)
  __shared__ float        sS[64][65];    // scores
  __shared__ unsigned int sP[64][36];    // probabilities, bf16 packed

  int bid  = blockIdx.x;           // widx*NH + h
  int h    = bid & 7;
  int widx = bid >> 3;
  int win  = widx & 63;
  int tid = threadIdx.x;
  int wave = tid >> 5, lane = tid & 31, lr = lane & 15, hi = lane >> 4;
  const float* bmt = bm + (size_t)(win * NH_ + h) * 4096;  // L2-resident tile

  // ---- load q,k,v; clamp padded rows (harmless: masked/never stored) ----
  {
    int r = tid >> 2, q = tid & 3;
    int rc = min(r, N_ - 1);
    size_t base = ((size_t)widx * N_ + rc) * 768 + h * 32 + q * 8;
    const float4* qp = (const float4*)(qkv + base);
    const float4* kp = (const float4*)(qkv + base + 256);
    const float4* vp = (const float4*)(qkv + base + 512);
    float4 q0 = qp[0], q1 = qp[1];
    float4 k0 = kp[0], k1 = kp[1];
    float4 v0 = vp[0], v1 = vp[1];
    uint4 uq = { packbf2(q0.x, q0.y), packbf2(q0.z, q0.w),
                 packbf2(q1.x, q1.y), packbf2(q1.z, q1.w) };
    uint4 uk = { packbf2(k0.x, k0.y), packbf2(k0.z, k0.w),
                 packbf2(k1.x, k1.y), packbf2(k1.z, k1.w) };
    *(uint4*)&sq[r][q * 4] = uq;
    *(uint4*)&sk[r][q * 4] = uk;
    unsigned short* svs = (unsigned short*)&svT[0][0];  // row stride 72 ushorts
    float vv[8] = { v0.x, v0.y, v0.z, v0.w, v1.x, v1.y, v1.z, v1.w };
    #pragma unroll
    for (int i = 0; i < 8; ++i)
      svs[(q * 8 + i) * 72 + r] = f2bfbits(vv[i]);
  }
  __syncthreads();

  // ---- Stage 1: S = Q K^T * scale + bias/mask table, 16 tiles ----
  #pragma unroll
  for (int tt = 0; tt < 2; ++tt) {
    int t  = wave + tt * 8;
    int qt = t >> 2, kt = t & 3;
    v16bf a = frag_ld(&sq[qt * 16 + lr][0], hi * 4, 8 + hi * 4);
    v16bf b = frag_ld(&sk[kt * 16 + lr][0], hi * 4, 8 + hi * 4);
    v8f c = {};
    c = __builtin_amdgcn_wmma_f32_16x16x32_bf16(false, a, false, b, (short)0, c, false, false);
    #pragma unroll
    for (int r = 0; r < 8; ++r) {
      int m = qt * 16 + r + 8 * hi;
      int n = kt * 16 + lr;
      sS[m][n] = fmaf(c[r], SCALE_, bmt[m * 64 + n]);
    }
  }
  __syncthreads();

  // ---- Stage 2: row softmax; pack probabilities to bf16 ----
  if (tid < 64) {
    float mx = -1e30f;
    for (int j = 0; j < 64; ++j) mx = fmaxf(mx, sS[tid][j]);
    float sum = 0.f;
    for (int j = 0; j < 64; ++j) { float e = __expf(sS[tid][j] - mx); sS[tid][j] = e; sum += e; }
    float inv = 1.0f / sum;
    for (int j = 0; j < 64; j += 2)
      sP[tid][j >> 1] = packbf2(sS[tid][j] * inv, sS[tid][j + 1] * inv);
  }
  __syncthreads();

  // ---- Stage 3: O = P V, 8 output tiles (4 qt x 2 dt), K=64 (2 WMMA) ----
  {
    int qt = wave >> 1, dt = wave & 1;
    v8f c = {};
    #pragma unroll
    for (int kc = 0; kc < 2; ++kc) {
      v16bf a = frag_ld(&sP[qt * 16 + lr][0],  kc * 16 + hi * 4, kc * 16 + 8 + hi * 4);
      v16bf b = frag_ld(&svT[dt * 16 + lr][0], kc * 16 + hi * 4, kc * 16 + 8 + hi * 4);
      c = __builtin_amdgcn_wmma_f32_16x16x32_bf16(false, a, false, b, (short)0, c, false, false);
    }
    #pragma unroll
    for (int r = 0; r < 8; ++r) {
      int m = qt * 16 + r + 8 * hi;
      if (m < N_)
        outbuf[((size_t)widx * N_ + m) * C_ + h * 32 + dt * 16 + lr] = c[r];
    }
  }
}

// ---------------------------------------------------------------------------
// Post-attention helpers
// ---------------------------------------------------------------------------
__global__ void spa_o_kernel(const float* __restrict__ projbuf, float* __restrict__ spa_o)
{
  int bp = blockIdx.x;             // b*P + p
  int bb = bp / P_, p = bp % P_;
  int c = threadIdx.x;
  float s = 0.f;
  for (int w = 0; w < NWIN_; ++w)
    s += projbuf[((size_t)(bb * NWIN_ + w) * N_ + p) * C_ + c];
  spa_o[(size_t)bp * C_ + c] = s * (1.0f / 64.0f);
}

__global__ void unwindow_kernel(const float* __restrict__ projbuf, float* __restrict__ xo)
{
  int bid = blockIdx.x;            // B*NWIN*49
  int t   = bid % NTOK_; int rem = bid / NTOK_;
  int win = rem % NWIN_; int bb  = rem / NWIN_;
  int wh = win >> 3, ww = win & 7;
  int th = t / WS_, tw = t % WS_;
  int hd = (wh * WS_ + th + 3) % H_;   // roll back by +3
  int wd = (ww * WS_ + tw + 3) % W_;
  int l  = hd * W_ + wd;
  int c = threadIdx.x;
  xo[((size_t)bb * L_ + l) * C_ + c] =
      projbuf[((size_t)(bb * NWIN_ + win) * N_ + P_ + t) * C_ + c];
}

__global__ __launch_bounds__(256)
void transpose_kernel(const float* __restrict__ xo, float* __restrict__ xot)
{
  __shared__ float tile[32][33];
  int bb = blockIdx.z;
  int l0 = blockIdx.x * 32;
  int c0 = blockIdx.y * 32;
  int tx = threadIdx.x & 31;
  int ty = threadIdx.x >> 5;       // 0..7
  #pragma unroll
  for (int i = 0; i < 4; ++i) {
    int ly = ty + i * 8;
    tile[ly][tx] = xo[((size_t)bb * L_ + l0 + ly) * C_ + c0 + tx];
  }
  __syncthreads();
  #pragma unroll
  for (int i = 0; i < 4; ++i) {
    int cy = ty + i * 8;
    xot[((size_t)bb * C_ + c0 + cy) * L_ + l0 + tx] = tile[tx][cy];
  }
}

__global__ void add2_kernel(const float* __restrict__ a, const float* __restrict__ b,
                            float* __restrict__ o, size_t n)
{
  size_t i = (size_t)blockIdx.x * blockDim.x + threadIdx.x;
  if (i < n) o[i] = a[i] + b[i];
}

__global__ void add3_kernel(const float* __restrict__ a, const float* __restrict__ b,
                            const float* __restrict__ c, float* __restrict__ o, size_t n)
{
  size_t i = (size_t)blockIdx.x * blockDim.x + threadIdx.x;
  if (i < n) o[i] = a[i] + b[i] + c[i];
}

// ---------------------------------------------------------------------------
// Channel attention: prompts query channels. One block per (b, head).
// ---------------------------------------------------------------------------
__global__ __launch_bounds__(256)
void chan_attn_kernel(const float* __restrict__ cq, const float* __restrict__ kvbuf,
                      float* __restrict__ co)
{
  __shared__ float scq[P_][CHD_];
  __shared__ float sca[P_][256];
  __shared__ float red[256];
  int bid = blockIdx.x;            // b*CH + h
  int h = bid & 3, bb = bid >> 2;
  int tid = threadIdx.x;

  for (int i = tid; i < P_ * CHD_; i += 256) {
    int p = i / CHD_, d = i % CHD_;
    scq[p][d] = cq[(size_t)(bb * P_ + p) * CE_ + h * CHD_ + d];
  }
  __syncthreads();

  {
    int c = tid;                   // channel 0..255
    float dot[P_] = {0, 0, 0, 0, 0};
    size_t base = ((size_t)(bb * 256 + c)) * 512 + h * CHD_;
    for (int d = 0; d < CHD_; ++d) {
      float kv = kvbuf[base + d];
      #pragma unroll
      for (int p = 0; p < P_; ++p) dot[p] += scq[p][d] * kv;
    }
    #pragma unroll
    for (int p = 0; p < P_; ++p) sca[p][c] = dot[p] * CSCALE_;
  }
  __syncthreads();

  for (int p = 0; p < P_; ++p) {
    red[tid] = sca[p][tid]; __syncthreads();
    for (int s = 128; s > 0; s >>= 1) { if (tid < s) red[tid] = fmaxf(red[tid], red[tid + s]); __syncthreads(); }
    float mx = red[0]; __syncthreads();
    float e = __expf(sca[p][tid] - mx);
    red[tid] = e; __syncthreads();
    for (int s = 128; s > 0; s >>= 1) { if (tid < s) red[tid] += red[tid + s]; __syncthreads(); }
    float sum = red[0]; __syncthreads();
    sca[p][tid] = e / sum;
    __syncthreads();
  }

  if (tid < CHD_) {
    int d = tid;
    float acc[P_] = {0, 0, 0, 0, 0};
    for (int c = 0; c < 256; ++c) {
      float vv = kvbuf[((size_t)(bb * 256 + c)) * 512 + 256 + h * CHD_ + d];
      #pragma unroll
      for (int p = 0; p < P_; ++p) acc[p] += sca[p][c] * vv;
    }
    #pragma unroll
    for (int p = 0; p < P_; ++p)
      co[(size_t)(bb * P_ + p) * CE_ + h * CHD_ + d] = acc[p];
  }
}

// ---------------------------------------------------------------------------
// Launch
// ---------------------------------------------------------------------------
extern "C" void kernel_launch(void* const* d_in, const int* in_sizes, int n_in,
                              void* d_out, int out_size, void* d_ws, size_t ws_size,
                              hipStream_t stream)
{
  (void)in_sizes; (void)n_in; (void)out_size; (void)ws_size;

  const float* x       = (const float*)d_in[0];
  const float* prompts = (const float*)d_in[1];
  const float* n1_g    = (const float*)d_in[2];
  const float* n1_b    = (const float*)d_in[3];
  const float* qkv_w   = (const float*)d_in[4];
  const float* qkv_b   = (const float*)d_in[5];
  const float* proj_w  = (const float*)d_in[6];
  const float* proj_b  = (const float*)d_in[7];
  const float* rpb     = (const float*)d_in[8];
  const float* n2_g    = (const float*)d_in[9];
  const float* n2_b    = (const float*)d_in[10];
  const float* fc1_w   = (const float*)d_in[11];
  const float* fc1_b   = (const float*)d_in[12];
  const float* fc2_w   = (const float*)d_in[13];
  const float* fc2_b   = (const float*)d_in[14];
  const float* tt_w    = (const float*)d_in[15];
  const float* tt_b    = (const float*)d_in[16];
  const float* cq_w    = (const float*)d_in[17];
  const float* cq_b    = (const float*)d_in[18];
  const float* ckv_w   = (const float*)d_in[19];
  const float* ckv_b   = (const float*)d_in[20];
  const float* cproj_w = (const float*)d_in[21];
  const float* cproj_b = (const float*)d_in[22];
  const float* tt1_w   = (const float*)d_in[23];
  const float* tt1_b   = (const float*)d_in[24];

  float* out    = (float*)d_out;
  float* out_xr = out;                                   // (B, L, C)
  float* out_tp = out + (size_t)B_ * L_ * C_;            // (B, P, C)

  // workspace layout (floats)
  float* ws = (float*)d_ws;
  const size_t R0 = 0;                       // 102,760,448  xn -> qkv -> mlp hidden
  const size_t R1 = 102760448;               //  28,311,552  xc -> proj -> ln(xr)
  const size_t R2 = 131072000;               //  28,311,552  attn_out -> xo
  const size_t R3 = 159383552;               //  25,690,112  xo^T
  const size_t R4 = 185073664;               //   4,194,304  channel kv
  const size_t SM = 189267968;
  float* xn      = ws + R0;
  float* qkvbuf  = ws + R0;
  float* hbuf    = ws + R0;
  float* xc      = ws + R1;
  float* projbuf = ws + R1;
  float* lnxr    = ws + R1;
  float* attnout = ws + R2;
  float* xo      = ws + R2;
  float* xot     = ws + R3;
  float* kvbuf   = ws + R4;
  float* spa     = ws + SM;
  float* chan    = spa   + 40960;
  float* cqb     = chan  + 40960;
  float* cob     = cqb   + 40960;
  float* co2     = cob   + 40960;
  float* co3     = co2   + 40960;
  float* tp0     = co3   + 40960;
  float* tpln    = tp0   + 40960;
  float* tph     = tpln  + 40960;            // 160 x 1024
  float* spa_o   = tph   + 163840;
  float* bm      = spa_o + 40960;            // 64*8*64*64 = 2,097,152 (8 MB)

  const int MT = B_ * NWIN_ * N_;            // 110592 token rows (windows)
  const int ML = B_ * L_;                    // 100352 pixel rows
  const int MP = B_ * P_;                    // 160 prompt rows

  // 0. precompute fused relpos-bias + shift-mask + padding table (8 MB, L2)
  bias_mask_kernel<<<NWIN_ * NH_, 256, 0, stream>>>(rpb, bm);

  // 1. spa = LN(prompts); chan = prompts @ tt_w^T + tt_b
  ln_rows_kernel<<<MP, 256, 0, stream>>>(prompts, spa, n1_g, n1_b, MP);
  gemm_wmma_kernel<0, false><<<dim3(3, 2), 256, 0, stream>>>(
      prompts, tt_w, tt_b, nullptr, chan, MP, CE_, C_);

  // 2. xn = LN(x); build xc (shift + window + prompt rows)
  ln_rows_kernel<<<ML, 256, 0, stream>>>(x, xn, n1_g, n1_b, ML);
  window_gather_kernel<<<B_ * NWIN_ * NTOK_, 256, 0, stream>>>(xn, xc);
  fill_spa_kernel<<<B_ * NWIN_, 256, 0, stream>>>(spa, xc);

  // 3. qkv = xc @ qkv_w^T + qkv_b     (110592 x 256 -> 768)
  gemm_wmma_kernel<0, false><<<dim3(MT / 64, 768 / 128), 256, 0, stream>>>(
      xc, qkv_w, qkv_b, nullptr, qkvbuf, MT, 768, C_);

  // 4. windowed attention (per window, head)
  attn_kernel<<<B_ * NWIN_ * NH_, 256, 0, stream>>>(qkvbuf, bm, attnout);

  // 5. proj
  gemm_wmma_kernel<0, false><<<dim3(MT / 64, C_ / 128), 256, 0, stream>>>(
      attnout, proj_w, proj_b, nullptr, projbuf, MT, C_, C_);

  // 6. spa_o (window mean of prompt rows), un-window + roll back, transpose
  spa_o_kernel<<<MP, 256, 0, stream>>>(projbuf, spa_o);
  unwindow_kernel<<<B_ * NWIN_ * NTOK_, 256, 0, stream>>>(projbuf, xo);
  transpose_kernel<<<dim3(L_ / 32, C_ / 32, B_), 256, 0, stream>>>(xo, xot);

  // 7. xr_pre = x + xo  -> out_xr
  {
    size_t n = (size_t)B_ * L_ * C_;
    add2_kernel<<<(unsigned)((n + 255) / 256), 256, 0, stream>>>(x, xo, out_xr, n);
  }

  // 8. channel attention
  gemm_wmma_kernel<0, false><<<dim3((B_ * C_) / 64, 512 / 128), 256, 0, stream>>>(
      xot, ckv_w, ckv_b, nullptr, kvbuf, B_ * C_, 512, L_);
  gemm_wmma_kernel<0, false><<<dim3(3, 2), 256, 0, stream>>>(
      chan, cq_w, cq_b, nullptr, cqb, MP, CE_, CE_);
  chan_attn_kernel<<<B_ * CH_, 256, 0, stream>>>(cqb, kvbuf, cob);
  gemm_wmma_kernel<0, false><<<dim3(3, 2), 256, 0, stream>>>(
      cob, cproj_w, cproj_b, nullptr, co2, MP, CE_, CE_);
  gemm_wmma_kernel<0, false><<<dim3(3, 2), 256, 0, stream>>>(
      co2, tt1_w, tt1_b, nullptr, co3, MP, C_, CE_);

  // 9. tp = prompts + spa_o + co3; tp += MLP(LN(tp))
  {
    size_t n = (size_t)MP * C_;
    add3_kernel<<<(unsigned)((n + 255) / 256), 256, 0, stream>>>(prompts, spa_o, co3, tp0, n);
  }
  ln_rows_kernel<<<MP, 256, 0, stream>>>(tp0, tpln, n2_g, n2_b, MP);
  gemm_wmma_kernel<1, false><<<dim3(3, 1024 / 128), 256, 0, stream>>>(
      tpln, fc1_w, fc1_b, nullptr, tph, MP, 1024, C_);
  gemm_wmma_kernel<0, true><<<dim3(3, 2), 256, 0, stream>>>(
      tph, fc2_w, fc2_b, tp0, out_tp, MP, C_, 1024);

  // 10. xr += MLP(LN(xr))
  ln_rows_kernel<<<ML, 256, 0, stream>>>(out_xr, lnxr, n2_g, n2_b, ML);
  gemm_wmma_kernel<1, false><<<dim3(ML / 64, 1024 / 128), 256, 0, stream>>>(
      lnxr, fc1_w, fc1_b, nullptr, hbuf, ML, 1024, C_);
  gemm_wmma_kernel<0, true><<<dim3(ML / 64, C_ / 128), 256, 0, stream>>>(
      hbuf, fc2_w, fc2_b, out_xr, out_xr, ML, C_, 1024);
}
